// BaseGPTNeo_81192061763683
// MI455X (gfx1250) — matrix operand order repaired and checked by reading, not compiled, not run on hardware
//
#include <hip/hip_runtime.h>
#include <hip/hip_bf16.h>
#include <stdint.h>

#define SEQ     2048
#define HIDDEN  1024
#define HEADS   16
#define HD      64
#define MLPDIM  4096
#define VOCAB   50257
#define LAYERS  6
#define WINDOW  256

typedef __bf16 v16bf __attribute__((ext_vector_type(16)));
typedef float  v8f   __attribute__((ext_vector_type(8)));
typedef unsigned int v4u __attribute__((ext_vector_type(4)));
typedef int          v8i __attribute__((ext_vector_type(8)));
typedef int          v4i __attribute__((ext_vector_type(4)));

// ---------------- small device helpers ----------------

__device__ __forceinline__ uint16_t f2bf(float f) {
    uint32_t u = __float_as_uint(f);
    u += 0x7fffu + ((u >> 16) & 1u);          // round-to-nearest-even bf16
    return (uint16_t)(u >> 16);
}

__device__ __forceinline__ uint32_t lds_off(const void* p) {
    // low 32 bits of a generic pointer into __shared__ = LDS byte offset
    return (uint32_t)(uintptr_t)p;
}

// gfx1250 async global->LDS copy (ASYNCcnt path).
__device__ __forceinline__ void async_ld_b128(void* lds, const void* g) {
    uint32_t l = lds_off(lds);
    unsigned long long a = (unsigned long long)(uintptr_t)g;
    asm volatile("global_load_async_to_lds_b128 %0, %1, off"
                 :: "v"(l), "v"(a) : "memory");
}
__device__ __forceinline__ void wait_async0() {
    asm volatile("s_wait_asynccnt 0x0" ::: "memory");
}

// gfx1250 Tensor Data Mover: 2D tile (tile_d0 x tile_d1 of 2-byte elements)
// from global (row stride stride_d0 elements) into LDS, with LDS row padding
// of 8 elements (16B) after every 64B row -> padded stride 40 elements.
// D# layout per CDNA5 ISA ch.8 (group0 128b, group1 256b).
// This toolchain exposes the 6-arg builtin form:
//   (uint32x4 g0, int32x8 g1, int32x4 g2, int32x4 g3, int32x8, i32 cpol)
__device__ __forceinline__ void tdm_load_2d(
    uint32_t lds_byte, const void* gaddr,
    uint32_t tile_d0, uint32_t tile_d1,
    uint32_t tensor_d0, uint32_t tensor_d1, uint32_t stride_d0) {
    const uint64_t ga = (uint64_t)(uintptr_t)gaddr;
    v4u g0;
    g0[0] = 1u;                                     // count=1, user descriptor
    g0[1] = lds_byte;                               // lds_addr [63:32]
    g0[2] = (uint32_t)ga;                           // global_addr low
    g0[3] = (uint32_t)(ga >> 32) | (2u << 30);      // global_addr high | type=2
    v8i g1;
    // data_size=2B (code 1); pad_enable, pad_interval=16 DW (code 3),
    // pad_amount=4 DW (code 3); workgroup_mask=0
    g1[0] = (int)((1u << 16) | (1u << 20) | (3u << 22) | (3u << 25));
    g1[1] = (int)((tensor_d0 & 0xffffu) << 16);     // [47:32]=0, tensor_dim0 lo
    g1[2] = (int)((tensor_d0 >> 16) | ((tensor_d1 & 0xffffu) << 16));
    g1[3] = (int)((tensor_d1 >> 16) | (tile_d0 << 16));
    g1[4] = (int)(tile_d1 & 0xffffu);               // tile_dim2 = 0
    g1[5] = (int)stride_d0;                         // tensor_dim0_stride lo
    g1[6] = 0;                                      // stride hi, dim1_stride lo
    g1[7] = 0;
    const v4i z4 = {0, 0, 0, 0};
    const v8i z8 = {0, 0, 0, 0, 0, 0, 0, 0};
    __builtin_amdgcn_tensor_load_to_lds(g0, g1, z4, z4, z8, 0);
}

__device__ __forceinline__ v8f wmma_bf16(v16bf a, v16bf b, v8f c) {
    return __builtin_amdgcn_wmma_f32_16x16x32_bf16(
        false, a, false, b, (short)0, c, false, false);
}

union FragU { uint4 u[2]; v16bf v; };

// A fragment (16x32 bf16): lane holds row r=lane&15; lanes 0-15: K 0..7 &
// 16..23, lanes 16-31: K 8..15 & 24..31.
__device__ __forceinline__ v16bf load_fragA(const uint16_t* base, int stride, int lane) {
    const int r = lane & 15, hh = lane >> 4;
    const uint16_t* p = base + r * stride + hh * 8;
    FragU f;
    f.u[0] = *(const uint4*)(p);
    f.u[1] = *(const uint4*)(p + 16);
    return f.v;
}
// B fragment (32x16 bf16) from a K-contiguous (transposed) tile:
// lane holds col n=lane&15; lanes 0-15: K 0..7 & 16..23, lanes 16-31: K 8..15 & 24..31.
__device__ __forceinline__ v16bf load_fragB(const uint16_t* base, int stride, int lane) {
    const int n = lane & 15, hh = lane >> 4;
    const uint16_t* p = base + n * stride + hh * 8;
    FragU f;
    f.u[0] = *(const uint4*)(p);
    f.u[1] = *(const uint4*)(p + 16);
    return f.v;
}

// ---------------- elementwise / prep kernels ----------------

__global__ __launch_bounds__(256) void embed_k(
    const int* __restrict__ ids, const float* __restrict__ wte,
    const float* __restrict__ wpe, float* __restrict__ x) {
    const int t = blockIdx.x;
    const int id = ids[t];
    for (int c = threadIdx.x; c < HIDDEN; c += 256)
        x[(size_t)t * HIDDEN + c] =
            wte[(size_t)id * HIDDEN + c] + wpe[(size_t)t * HIDDEN + c];
}

__global__ __launch_bounds__(256) void layernorm_bf16(
    const float* __restrict__ X, const float* __restrict__ w,
    const float* __restrict__ b, uint16_t* __restrict__ out) {
    const int t = blockIdx.x;
    const float* xr = X + (size_t)t * HIDDEN;
    float v[4]; float s = 0.f, s2 = 0.f;
#pragma unroll
    for (int i = 0; i < 4; ++i) {
        float f = xr[threadIdx.x + i * 256];
        v[i] = f; s += f; s2 += f * f;
    }
#pragma unroll
    for (int m = 16; m >= 1; m >>= 1) {
        s  += __shfl_xor(s,  m, 32);
        s2 += __shfl_xor(s2, m, 32);
    }
    __shared__ float rs[8], rs2[8];
    const int wave = threadIdx.x >> 5, lane = threadIdx.x & 31;
    if (lane == 0) { rs[wave] = s; rs2[wave] = s2; }
    __syncthreads();
    s = rs[lane & 7]; s2 = rs2[lane & 7];
#pragma unroll
    for (int m = 4; m >= 1; m >>= 1) {
        s  += __shfl_xor(s,  m, 32);
        s2 += __shfl_xor(s2, m, 32);
    }
    const float mean = s * (1.f / HIDDEN);
    const float var  = s2 * (1.f / HIDDEN) - mean * mean;
    const float rstd = rsqrtf(var + 1e-5f);
#pragma unroll
    for (int i = 0; i < 4; ++i) {
        const int c = threadIdx.x + i * 256;
        out[(size_t)t * HIDDEN + c] = f2bf((v[i] - mean) * rstd * w[c] + b[c]);
    }
}

// W fp32 [K][N]  ->  Wt bf16 [N][K]  (K-contiguous rows for WMMA B operand)
__global__ __launch_bounds__(256) void convert_transpose(
    const float* __restrict__ W, uint16_t* __restrict__ Wt, int K, int N) {
    __shared__ float tile[32][33];
    const int tx = threadIdx.x & 31, ty = threadIdx.x >> 5;
    const int n = blockIdx.x * 32 + tx;
#pragma unroll
    for (int r = 0; r < 32; r += 8) {
        const int k = blockIdx.y * 32 + ty + r;
        tile[ty + r][tx] = W[(size_t)k * N + n];
    }
    __syncthreads();
    const int k2 = blockIdx.y * 32 + tx;
#pragma unroll
    for (int r = 0; r < 32; r += 8) {
        const int n2 = blockIdx.x * 32 + ty + r;
        Wt[(size_t)n2 * K + k2] = f2bf(tile[tx][ty + r]);
    }
}

__global__ __launch_bounds__(256) void conv_bf16_k(
    const float* __restrict__ in, uint16_t* __restrict__ out, int n) {
    const int i = blockIdx.x * 256 + threadIdx.x;
    if (i < n) out[i] = f2bf(in[i]);
}

// ---------------- WMMA GEMM: C[M,N] = A[M,K](bf16) x Bt[N,K](bf16) ----------
// 128x128 block tile, 8 waves (4x2), each wave 32x64 (2x4 WMMA frags),
// K stepped by 32.  Tiles staged to LDS by the Tensor Data Mover (one
// tensor_load_to_lds descriptor per operand per k-step, issued by wave 0,
// TENSORcnt-synchronized), double-buffered.
// EPI: 0 = bf16 store, 1 = bf16 transposed store, 2 = fp32 +=acc+bias,
//      3 = bf16 gelu(acc+bias), 4 = fp32 store

template <int EPI>
__global__ __launch_bounds__(256) void gemm128(
    const uint16_t* __restrict__ A, int lda,
    const uint16_t* __restrict__ Bt, int ldb,
    float* __restrict__ Cf, uint16_t* __restrict__ Cb, int ldc,
    int N, int K, const float* __restrict__ bias) {
    __shared__ __align__(16) uint16_t sA[2][128 * 40];
    __shared__ __align__(16) uint16_t sB[2][128 * 40];
    const int tid = threadIdx.x, lane = tid & 31, wave = tid >> 5;
    const int m0 = blockIdx.y * 128, n0 = blockIdx.x * 128;
    const int wm = wave >> 1, wn = wave & 1;
    v8f acc[2][4] = {};

    const int nk = K >> 5;
    auto issue = [&](int kt, int buf) {
        if (wave == 0) {
            const int k0 = kt << 5;
            // A tile: 32 x 128 rows, OOB-safe (M multiple of 128)
            tdm_load_2d(lds_off(&sA[buf][0]),
                        A + (size_t)m0 * lda + k0,
                        32u, 128u, (uint32_t)K, 128u, (uint32_t)lda);
            // B tile: rows past (N - n0) read back zero via tensor_dim1 OOB
            tdm_load_2d(lds_off(&sB[buf][0]),
                        Bt + (size_t)n0 * ldb + k0,
                        32u, 128u, (uint32_t)K, (uint32_t)(N - n0), (uint32_t)ldb);
        }
    };

    issue(0, 0);
    for (int kt = 0; kt < nk; ++kt) {
        __builtin_amdgcn_s_wait_tensorcnt(0);
        __syncthreads();
        if (kt + 1 < nk) issue(kt + 1, (kt + 1) & 1);
        const uint16_t* Ab = &sA[kt & 1][(wm * 32) * 40];
        const uint16_t* Bb = &sB[kt & 1][(wn * 64) * 40];
        v16bf a0 = load_fragA(Ab,           40, lane);
        v16bf a1 = load_fragA(Ab + 16 * 40, 40, lane);
#pragma unroll
        for (int j = 0; j < 4; ++j) {
            v16bf bfr = load_fragB(Bb + j * 16 * 40, 40, lane);
            acc[0][j] = wmma_bf16(a0, bfr, acc[0][j]);
            acc[1][j] = wmma_bf16(a1, bfr, acc[1][j]);
        }
    }

    const int hh = lane >> 4, col = lane & 15;
#pragma unroll
    for (int i = 0; i < 2; ++i)
#pragma unroll
        for (int j = 0; j < 4; ++j)
#pragma unroll
            for (int e = 0; e < 8; ++e) {
                const int m = m0 + wm * 32 + i * 16 + hh * 8 + e;
                const int n = n0 + wn * 64 + j * 16 + col;
                if (n < N) {
                    const float v = acc[i][j][e];
                    if constexpr (EPI == 0) {
                        Cb[(size_t)m * ldc + n] = f2bf(v);
                    } else if constexpr (EPI == 1) {
                        Cb[(size_t)n * ldc + m] = f2bf(v);
                    } else if constexpr (EPI == 2) {
                        Cf[(size_t)m * ldc + n] += v + bias[n];
                    } else if constexpr (EPI == 3) {
                        const float t = v + bias[n];
                        Cb[(size_t)m * ldc + n] =
                            f2bf(0.5f * t * (1.0f + erff(t * 0.70710678118f)));
                    } else {
                        Cf[(size_t)m * ldc + n] = v;
                    }
                }
            }
}

// ---------------- flash attention (no 1/sqrt(d) scale, GPT-Neo) -------------
// One block = 128 queries of one head; 8 waves x 16 query rows.
// 64-key tiles staged in LDS via async loads; S and P*V through bf16 WMMA.

__global__ __launch_bounds__(256) void attn_fa(
    const uint16_t* __restrict__ Q, const uint16_t* __restrict__ Kb,
    const uint16_t* __restrict__ VT, uint16_t* __restrict__ CTX, int is_local) {
    __shared__ __align__(16) uint16_t s_k[64 * 72];      // [key][d]
    __shared__ __align__(16) uint16_t s_v[64 * 72];      // [d][key]
    __shared__ __align__(16) uint16_t s_p[8 * 16 * 72];  // per-wave P staging
    const int tid = threadIdx.x, lane = tid & 31, wave = tid >> 5;
    const int qb = blockIdx.x, head = blockIdx.y;
    const int q0 = qb * 128, hoff = head * HD;
    const int hh = lane >> 4, col = lane & 15;

    // Q fragments for this wave's 16 rows, straight from global (bf16)
    const uint16_t* qbase = Q + (size_t)(q0 + wave * 16) * HIDDEN + hoff;
    const v16bf a0 = load_fragA(qbase,      HIDDEN, lane);   // d 0..31
    const v16bf a1 = load_fragA(qbase + 32, HIDDEN, lane);   // d 32..63

    float m_i[8], l_i[8];
    v8f o[4] = {};
#pragma unroll
    for (int i = 0; i < 8; ++i) { m_i[i] = -3.0e38f; l_i[i] = 0.f; }

    const int kb_end = 2 * qb + 1;
    int kb_start = 0;
    if (is_local) { kb_start = 2 * qb - 4; if (kb_start < 0) kb_start = 0; }
    uint16_t* pbase = s_p + wave * 16 * 72;

    for (int kb = kb_start; kb <= kb_end; ++kb) {
        const int kcol0 = kb * 64;
        for (int i = tid; i < 512; i += 256) {          // K tile: 64 keys x 8 segs
            const int r = i >> 3, s = i & 7;
            async_ld_b128(&s_k[r * 72 + s * 8],
                          Kb + (size_t)(kcol0 + r) * HIDDEN + hoff + s * 8);
        }
        for (int i = tid; i < 512; i += 256) {          // V^T tile: 64 d x 8 segs
            const int d = i >> 3, s = i & 7;
            async_ld_b128(&s_v[d * 72 + s * 8],
                          VT + (size_t)(hoff + d) * SEQ + kcol0 + s * 8);
        }
        wait_async0();
        __syncthreads();

        // S = Q(16x64) @ K^T -> 4 frags of 16x16 over 64 keys
        v8f sfr[4];
#pragma unroll
        for (int f = 0; f < 4; ++f) {
            v16bf b0 = load_fragB(s_k + (f * 16) * 72,      72, lane);
            v16bf b1 = load_fragB(s_k + (f * 16) * 72 + 32, 72, lane);
            v8f z = {};
            z = wmma_bf16(a0, b0, z);
            sfr[f] = wmma_bf16(a1, b1, z);
        }

        // online softmax, 8 rows per lane-half
#pragma unroll
        for (int i = 0; i < 8; ++i) {
            const int row_q = q0 + wave * 16 + hh * 8 + i;
            float mx = -3.0e38f;
#pragma unroll
            for (int f = 0; f < 4; ++f) {
                const int j = kcol0 + f * 16 + col;
                const bool ok = (j <= row_q) && (!is_local || (row_q - j) < WINDOW);
                const float sv = ok ? sfr[f][i] : -3.0e38f;
                sfr[f][i] = sv;
                mx = fmaxf(mx, sv);
            }
#pragma unroll
            for (int msk = 8; msk >= 1; msk >>= 1)
                mx = fmaxf(mx, __shfl_xor(mx, msk, 32));
            const float mnew  = fmaxf(m_i[i], mx);
            const float scale = __expf(m_i[i] - mnew);
            float lsum = 0.f;
#pragma unroll
            for (int f = 0; f < 4; ++f) {
                const float p = __expf(sfr[f][i] - mnew);
                pbase[(hh * 8 + i) * 72 + f * 16 + col] = f2bf(p);
                lsum += p;
            }
#pragma unroll
            for (int msk = 8; msk >= 1; msk >>= 1)
                lsum += __shfl_xor(lsum, msk, 32);
            l_i[i] = l_i[i] * scale + lsum;
            m_i[i] = mnew;
#pragma unroll
            for (int fd = 0; fd < 4; ++fd) o[fd][i] *= scale;
        }

        // O += P(16x64) @ V(64x64): A = per-wave P (LDS in-order same-wave)
#pragma unroll
        for (int kc = 0; kc < 2; ++kc) {
            v16bf ap = load_fragA(pbase + kc * 32, 72, lane);
#pragma unroll
            for (int fd = 0; fd < 4; ++fd) {
                v16bf bv = load_fragB(s_v + (fd * 16) * 72 + kc * 32, 72, lane);
                o[fd] = wmma_bf16(ap, bv, o[fd]);
            }
        }
        __syncthreads();   // protect s_k/s_v before next tile's async writes
    }

#pragma unroll
    for (int i = 0; i < 8; ++i) {
        const float inv = 1.0f / l_i[i];
        const int row_q = q0 + wave * 16 + hh * 8 + i;
#pragma unroll
        for (int fd = 0; fd < 4; ++fd)
            CTX[(size_t)row_q * HIDDEN + hoff + fd * 16 + col] =
                f2bf(o[fd][i] * inv);
    }
}

// ---------------- host orchestration ----------------

extern "C" void kernel_launch(void* const* d_in, const int* in_sizes, int n_in,
                              void* d_out, int out_size, void* d_ws, size_t ws_size,
                              hipStream_t stream) {
    (void)in_sizes; (void)n_in; (void)out_size; (void)ws_size;
    const int*   ids    = (const int*)  d_in[0];
    const float* wte    = (const float*)d_in[1];
    const float* wpe    = (const float*)d_in[2];
    const float* ln1_w  = (const float*)d_in[3];
    const float* ln1_b  = (const float*)d_in[4];
    const float* q_w    = (const float*)d_in[5];
    const float* k_w    = (const float*)d_in[6];
    const float* v_w    = (const float*)d_in[7];
    const float* o_w    = (const float*)d_in[8];
    const float* o_b    = (const float*)d_in[9];
    const float* ln2_w  = (const float*)d_in[10];
    const float* ln2_b  = (const float*)d_in[11];
    const float* fc_w   = (const float*)d_in[12];
    const float* fc_b   = (const float*)d_in[13];
    const float* proj_w = (const float*)d_in[14];
    const float* proj_b = (const float*)d_in[15];
    const float* lnf_w  = (const float*)d_in[16];
    const float* lnf_b  = (const float*)d_in[17];
    float* out = (float*)d_out;

    char* ws = (char*)d_ws;
    size_t off = 0;
    auto wsalloc = [&](size_t bytes) {
        void* p = ws + off;
        off += (bytes + 255) & ~(size_t)255;
        return p;
    };
    float*    x   = (float*)   wsalloc((size_t)SEQ * HIDDEN * 4);  // residual fp32
    uint16_t* h   = (uint16_t*)wsalloc((size_t)SEQ * HIDDEN * 2);  // LN out bf16
    uint16_t* qB  = (uint16_t*)wsalloc((size_t)SEQ * HIDDEN * 2);
    uint16_t* kB  = (uint16_t*)wsalloc((size_t)SEQ * HIDDEN * 2);
    uint16_t* vT  = (uint16_t*)wsalloc((size_t)SEQ * HIDDEN * 2);  // [HIDDEN][SEQ]
    uint16_t* ctx = (uint16_t*)wsalloc((size_t)SEQ * HIDDEN * 2);
    uint16_t* mid = (uint16_t*)wsalloc((size_t)SEQ * MLPDIM * 2);
    uint16_t* wt  = (uint16_t*)wsalloc((size_t)8192 * HIDDEN * 2); // weight staging

    const dim3 blk(256);
    embed_k<<<SEQ, blk, 0, stream>>>(ids, wte, wpe, x);

    for (int l = 0; l < LAYERS; ++l) {
        const int is_local = l & 1;   // ["global","local",...]
        layernorm_bf16<<<SEQ, blk, 0, stream>>>(x, ln1_w + l * HIDDEN, ln1_b + l * HIDDEN, h);

        // Q = h @ q_w
        convert_transpose<<<dim3(HIDDEN / 32, HIDDEN / 32), blk, 0, stream>>>(
            q_w + (size_t)l * HIDDEN * HIDDEN, wt, HIDDEN, HIDDEN);
        gemm128<0><<<dim3(HIDDEN / 128, SEQ / 128), blk, 0, stream>>>(
            h, HIDDEN, wt, HIDDEN, nullptr, qB, HIDDEN, HIDDEN, HIDDEN, nullptr);
        // K = h @ k_w
        convert_transpose<<<dim3(HIDDEN / 32, HIDDEN / 32), blk, 0, stream>>>(
            k_w + (size_t)l * HIDDEN * HIDDEN, wt, HIDDEN, HIDDEN);
        gemm128<0><<<dim3(HIDDEN / 128, SEQ / 128), blk, 0, stream>>>(
            h, HIDDEN, wt, HIDDEN, nullptr, kB, HIDDEN, HIDDEN, HIDDEN, nullptr);
        // V^T = (h @ v_w)^T  (transposed store so P*V B-operand is K-contiguous)
        convert_transpose<<<dim3(HIDDEN / 32, HIDDEN / 32), blk, 0, stream>>>(
            v_w + (size_t)l * HIDDEN * HIDDEN, wt, HIDDEN, HIDDEN);
        gemm128<1><<<dim3(HIDDEN / 128, SEQ / 128), blk, 0, stream>>>(
            h, HIDDEN, wt, HIDDEN, nullptr, vT, SEQ, HIDDEN, HIDDEN, nullptr);

        attn_fa<<<dim3(SEQ / 128, HEADS), blk, 0, stream>>>(qB, kB, vT, ctx, is_local);

        // x += ctx @ o_w + o_b
        convert_transpose<<<dim3(HIDDEN / 32, HIDDEN / 32), blk, 0, stream>>>(
            o_w + (size_t)l * HIDDEN * HIDDEN, wt, HIDDEN, HIDDEN);
        gemm128<2><<<dim3(HIDDEN / 128, SEQ / 128), blk, 0, stream>>>(
            ctx, HIDDEN, wt, HIDDEN, x, nullptr, HIDDEN, HIDDEN, HIDDEN,
            o_b + (size_t)l * HIDDEN);

        layernorm_bf16<<<SEQ, blk, 0, stream>>>(x, ln2_w + l * HIDDEN, ln2_b + l * HIDDEN, h);

        // mid = gelu(h @ fc_w + fc_b)
        convert_transpose<<<dim3(MLPDIM / 32, HIDDEN / 32), blk, 0, stream>>>(
            fc_w + (size_t)l * HIDDEN * MLPDIM, wt, HIDDEN, MLPDIM);
        gemm128<3><<<dim3(MLPDIM / 128, SEQ / 128), blk, 0, stream>>>(
            h, HIDDEN, wt, HIDDEN, nullptr, mid, MLPDIM, MLPDIM, HIDDEN,
            fc_b + (size_t)l * MLPDIM);

        // x += mid @ proj_w + proj_b
        convert_transpose<<<dim3(HIDDEN / 32, MLPDIM / 32), blk, 0, stream>>>(
            proj_w + (size_t)l * MLPDIM * HIDDEN, wt, MLPDIM, HIDDEN);
        gemm128<2><<<dim3(HIDDEN / 128, SEQ / 128), blk, 0, stream>>>(
            mid, MLPDIM, wt, MLPDIM, x, nullptr, HIDDEN, HIDDEN, MLPDIM,
            proj_b + (size_t)l * HIDDEN);
    }

    layernorm_bf16<<<SEQ, blk, 0, stream>>>(x, lnf_w, lnf_b, h);

    // logits = h @ wte^T, chunked over vocab through the bf16 staging buffer
    for (int n0 = 0; n0 < VOCAB; n0 += 8192) {
        int chunk = VOCAB - n0; if (chunk > 8192) chunk = 8192;
        const int nelem = chunk * HIDDEN;
        conv_bf16_k<<<(nelem + 255) / 256, blk, 0, stream>>>(
            wte + (size_t)n0 * HIDDEN, wt, nelem);
        gemm128<4><<<dim3((chunk + 127) / 128, SEQ / 128), blk, 0, stream>>>(
            h, HIDDEN, wt, HIDDEN, out + n0, nullptr, VOCAB, chunk, HIDDEN, nullptr);
    }
}
